// ShiTomasiSparseBADSinkhornMatcher_43026982371761
// MI455X (gfx1250) — matrix operand order repaired and checked by reading, not compiled, not run on hardware
//
#include <hip/hip_runtime.h>
#include <math.h>

typedef __attribute__((ext_vector_type(16))) _Float16 v16h;
typedef __attribute__((ext_vector_type(8)))  float    v8f;

#define BN 4
#define IH 1024
#define IW 1024
#define KK 2048      // keypoints per image
#define PP 256       // descriptor dim
#define MAXR 16
#define NMSR 3
#define NAUG 2049
#define SINK_ITERS 20

__device__ __forceinline__ int iclamp(int v, int lo, int hi){
  return v < lo ? lo : (v > hi ? hi : v);
}
__device__ __forceinline__ float px(const float* im, int y, int x){
  return (y >= 0 && y < IH && x >= 0 && x < IW) ? im[y*IW + x] : 0.f;
}

// ---------------- Phase 1: Shi-Tomasi score --------------------------------
__global__ void k_shitomasi(const float* __restrict__ img, float* __restrict__ s){
  int gid = blockIdx.x*blockDim.x + threadIdx.x;
  if (gid >= BN*IH*IW) return;
  int b = gid / (IH*IW);
  int rem = gid - b*(IH*IW);
  int y = rem / IW, x = rem - y*IW;
  const float* im = img + (size_t)b*IH*IW;
  float a = 0.f, bb = 0.f, c = 0.f;
  #pragma unroll
  for (int dy = -1; dy <= 1; ++dy){
    #pragma unroll
    for (int dx = -1; dx <= 1; ++dx){
      int yy = y + dy, xx = x + dx;
      if (yy < 0 || yy >= IH || xx < 0 || xx >= IW) continue; // zero-padded box conv
      float p00 = px(im, yy-1, xx-1), p01 = px(im, yy-1, xx), p02 = px(im, yy-1, xx+1);
      float p10 = px(im, yy,   xx-1),                          p12 = px(im, yy,   xx+1);
      float p20 = px(im, yy+1, xx-1), p21 = px(im, yy+1, xx), p22 = px(im, yy+1, xx+1);
      float ix = (p02 - p00) + 2.f*(p12 - p10) + (p22 - p20);
      float iy = (p20 - p00) + 2.f*(p21 - p01) + (p22 - p02);
      a += ix*ix; bb += iy*iy; c += ix*iy;
    }
  }
  float root = sqrtf(fmaxf((a - bb)*(a - bb) + 4.f*c*c, 0.f));
  s[gid] = 0.5f*(a + bb - root);
}

// ---------------- Phase 2: NMS + border + clip -----------------------------
__global__ void k_nms(const float* __restrict__ s, float* __restrict__ sm){
  int gid = blockIdx.x*blockDim.x + threadIdx.x;
  if (gid >= BN*IH*IW) return;
  int b = gid / (IH*IW);
  int rem = gid - b*(IH*IW);
  int y = rem / IW, x = rem - y*IW;
  const float* sb = s + (size_t)b*IH*IW;
  float v0 = sb[y*IW + x];
  float lm = -INFINITY;
  for (int dy = -NMSR; dy <= NMSR; ++dy){
    int yy = y + dy; if (yy < 0 || yy >= IH) continue;
    for (int dx = -NMSR; dx <= NMSR; ++dx){
      int xx = x + dx; if (xx < 0 || xx >= IW) continue;
      lm = fmaxf(lm, sb[yy*IW + xx]);
    }
  }
  float m = (v0 >= lm - 1e-7f) ? 1.f : 0.f;
  float border = (y >= MAXR && y < IH-MAXR && x >= MAXR && x < IW-MAXR) ? 1.f : 0.f;
  float val = v0 * m * border;
  sm[gid] = (val > 0.f) ? val : 0.f;
}

// ---------------- Phase 3: top-K via radix select + bitonic sort -----------
__global__ void k_zero(unsigned* __restrict__ p, int n){
  int i = blockIdx.x*blockDim.x + threadIdx.x;
  if (i < n) p[i] = 0u;
}

__global__ void k_hist(const float* __restrict__ sm, unsigned* __restrict__ hist){
  int gid = blockIdx.x*blockDim.x + threadIdx.x;
  if (gid >= BN*IH*IW) return;
  int b = gid / (IH*IW);
  float v = sm[gid];
  if (v > 0.f){
    unsigned bin = __float_as_uint(v) >> 16;   // positive floats: bits order == value order
    atomicAdd(&hist[b*65536u + bin], 1u);
  }
}

__global__ void k_select(const unsigned* __restrict__ hist, int* __restrict__ sel){
  if (threadIdx.x != 0) return;
  int b = blockIdx.x;
  const unsigned* h = hist + (size_t)b*65536;
  int cum = 0, T = -1, slots = 0;
  for (int bin = 65535; bin >= 0; --bin){
    int c = (int)h[bin];
    if (cum + c >= KK){ T = bin; slots = KK - cum; break; }
    cum += c;
  }
  sel[2*b] = T; sel[2*b+1] = slots;   // T=-1 -> fewer than K positives, take all
}

__global__ void k_compact(const float* __restrict__ sm, const int* __restrict__ sel,
                          unsigned* __restrict__ cnt, unsigned* __restrict__ cntT,
                          float* __restrict__ pairSc, int* __restrict__ pairIdx){
  int gid = blockIdx.x*blockDim.x + threadIdx.x;
  if (gid >= BN*IH*IW) return;
  int b = gid / (IH*IW);
  int rem = gid - b*(IH*IW);
  float v = sm[gid];
  if (v <= 0.f) return;
  int T = sel[2*b], slots = sel[2*b+1];
  int bin = (int)(__float_as_uint(v) >> 16);
  if (bin > T){
    unsigned pos = atomicAdd(&cnt[b], 1u);
    if (pos < KK){ pairSc[b*KK + pos] = v; pairIdx[b*KK + pos] = rem; }
  } else if (bin == T){
    unsigned tp = atomicAdd(&cntT[b], 1u);
    if ((int)tp < slots){
      unsigned pos = atomicAdd(&cnt[b], 1u);
      if (pos < KK){ pairSc[b*KK + pos] = v; pairIdx[b*KK + pos] = rem; }
    }
  }
}

__global__ __launch_bounds__(1024)
void k_sortemit(const float* __restrict__ pairSc, const int* __restrict__ pairIdx,
                const unsigned* __restrict__ cnt, float* __restrict__ outKp,
                float* __restrict__ validBuf){
  __shared__ float sc[KK];
  __shared__ int   id[KK];
  int b = blockIdx.x, t = threadIdx.x;
  int n = (int)cnt[b]; if (n > KK) n = KK;
  for (int i = t; i < KK; i += 1024){
    if (i < n){ sc[i] = pairSc[b*KK + i]; id[i] = pairIdx[b*KK + i]; }
    else      { sc[i] = -1.f;             id[i] = 0x7FFFFFFF; }
  }
  __syncthreads();
  for (int k = 2; k <= KK; k <<= 1){
    for (int j = k >> 1; j > 0; j >>= 1){
      for (int i = t; i < KK; i += 1024){
        int l = i ^ j;
        if (l > i){
          float si = sc[i], sl = sc[l]; int ii = id[i], il = id[l];
          bool iBeforeL = (si > sl) || (si == sl && ii < il); // descending, tie: low idx
          bool desc = ((i & k) == 0);
          bool doSwap = desc ? !iBeforeL : iBeforeL;
          if (doSwap){ sc[i] = sl; sc[l] = si; id[i] = il; id[l] = ii; }
        }
      }
      __syncthreads();
    }
  }
  for (int i = t; i < KK; i += 1024){
    float v = sc[i];
    bool ok = v > 0.f;
    int idx = id[i];
    outKp[(b*KK + i)*2 + 0] = ok ? (float)(idx / IW) : -1.f;
    outKp[(b*KK + i)*2 + 1] = ok ? (float)(idx % IW) : -1.f;
    validBuf[b*KK + i] = ok ? 1.f : 0.f;
  }
}

// ---------------- Phase 4: sparse BAD descriptors (fp16 output) ------------
__global__ void k_desc(const float* __restrict__ img, const float* __restrict__ kp,
                       const float* __restrict__ validBuf,
                       const float* __restrict__ offsA, const float* __restrict__ offsB,
                       _Float16* __restrict__ dOut, float* __restrict__ sqOut){
  int kpIdx = blockIdx.x;              // 0..BN*KK-1
  int b = kpIdx / KK;
  int p = threadIdx.x;                 // 0..255
  const float* im = img + (size_t)b*IH*IW;
  float ky = kp[kpIdx*2 + 0], kx = kp[kpIdx*2 + 1];
  float vld = validBuf[kpIdx];
  float ya = ky + offsA[p*2+0], xa = kx + offsA[p*2+1];
  float yb = ky + offsB[p*2+0], xb = kx + offsB[p*2+1];
  int yai = iclamp((int)rintf(ya), 0, IH-1);
  int xai = iclamp((int)rintf(xa), 0, IW-1);
  int ybi = iclamp((int)rintf(yb), 0, IH-1);
  int xbi = iclamp((int)rintf(xb), 0, IW-1);
  float d = (im[yai*IW + xai] - im[ybi*IW + xbi]) * vld;

  __shared__ float red[PP];
  red[p] = d*d; __syncthreads();
  for (int sft = PP/2; sft > 0; sft >>= 1){
    if (p < sft) red[p] += red[p + sft];
    __syncthreads();
  }
  float s2 = red[0];
  float den = sqrtf(s2) + 1e-8f;
  float dn = d / den;
  dOut[(size_t)kpIdx*PP + p] = (_Float16)dn;
  if (p == 0) sqOut[kpIdx] = s2 / (den*den);
}

// ---------------- Phase 5: augmented-Z init + WMMA GEMM --------------------
__global__ void k_auginit(float* __restrict__ Z){
  int j = blockIdx.x*blockDim.x + threadIdx.x;
  int b = blockIdx.y;
  if (j >= NAUG) return;
  float* Zb = Z + (size_t)b*NAUG*NAUG;
  Zb[(size_t)(NAUG-1)*NAUG + j] = 1.0f;   // bin row   (BIN_SCORE / EPS = 1)
  Zb[(size_t)j*NAUG + (NAUG-1)] = 1.0f;   // bin column
}

#define WMMA_F16(A, B, C) \
  __builtin_amdgcn_wmma_f32_16x16x32_f16(false, (A), false, (B), (short)0, (C), false, false)

// 2x2 register-blocked WMMA GEMM: each wave owns a 32x32 output tile
// (2 A-fragments x 2 B-fragments -> 4 independent accumulators per k-step,
//  each fragment feeds two WMMAs; 32 v_wmma_f32_16x16x32_f16 per wave).
// 4 waves per block, each on its own tile (EXEC all-ones for WMMA).
__global__ __launch_bounds__(128)
void k_gemm(const _Float16* __restrict__ d1, const _Float16* __restrict__ d2,
            const float* __restrict__ sq1, const float* __restrict__ sq2,
            float* __restrict__ Z){
  int wave = threadIdx.x >> 5;
  int l    = threadIdx.x & 31;          // lane, wave32
  int mt = blockIdx.x*4 + wave;         // 32-row tile index (0..63)
  int nt = blockIdx.y;                  // 32-col tile index (0..63)
  int b  = blockIdx.z;
  int m0 = mt*32, n0 = nt*32;
  int hi = l >> 4, lo = l & 15;
  const _Float16* A0 = d1 + ((size_t)(b*KK + m0 + lo))*PP;   // rows m0..m0+15
  const _Float16* A1 = A0 + (size_t)16*PP;                   // rows m0+16..m0+31
  const _Float16* B0 = d2 + ((size_t)(b*KK + n0 + lo))*PP;   // cols n0..n0+15
  const _Float16* B1 = B0 + (size_t)16*PP;                   // cols n0+16..n0+31
  v8f acc00 = {}, acc01 = {}, acc10 = {}, acc11 = {};
  #pragma unroll
  for (int kk = 0; kk < PP; kk += 32){
    v16h a0, a1, b0f, b1f;
    // 16-bit A 16x32 layout (ISA 7.12.2): lanes 0-15 K=0..7,16..23; lanes 16-31 +8
    int ka = kk + 8*hi;
    #pragma unroll
    for (int q = 0; q < 8; ++q){
      int k = (q < 4 ? 2*q : 2*q + 8) + ka;
      a0[2*q] = A0[k]; a0[2*q+1] = A0[k+1];
      a1[2*q] = A1[k]; a1[2*q+1] = A1[k+1];
    }
    // B 32x16: same K order as A: lanes 0-15 K=0..7,16..23; lanes 16-31 +8, N=lane&15
    int kb = kk + 8*hi;
    #pragma unroll
    for (int h = 0; h < 16; ++h){ int kh = kb + (h < 8 ? h : h + 8); b0f[h] = B0[kh]; b1f[h] = B1[kh]; }
    acc00 = WMMA_F16(a0, b0f, acc00);
    acc01 = WMMA_F16(a0, b1f, acc01);
    acc10 = WMMA_F16(a1, b0f, acc10);
    acc11 = WMMA_F16(a1, b1f, acc11);
  }
  // C/D layout: VGPR i -> M = tile_m + i + 8*hi, N = tile_n + (l&15)
  float sqn0 = sq2[b*KK + n0 + lo];
  float sqn1 = sq2[b*KK + n0 + 16 + lo];
  float* Zb = Z + (size_t)b*NAUG*NAUG;
  int nc0 = n0 + lo, nc1 = n0 + 16 + lo;
  #pragma unroll
  for (int i = 0; i < 8; ++i){
    int mr0 = m0 + i + 8*hi;
    int mr1 = mr0 + 16;
    float sqm0 = sq1[b*KK + mr0];
    float sqm1 = sq1[b*KK + mr1];
    Zb[(size_t)mr0*NAUG + nc0] = -fmaxf(sqm0 + sqn0 - 2.f*acc00[i], 0.f);
    Zb[(size_t)mr0*NAUG + nc1] = -fmaxf(sqm0 + sqn1 - 2.f*acc01[i], 0.f);
    Zb[(size_t)mr1*NAUG + nc0] = -fmaxf(sqm1 + sqn0 - 2.f*acc10[i], 0.f);
    Zb[(size_t)mr1*NAUG + nc1] = -fmaxf(sqm1 + sqn1 - 2.f*acc11[i], 0.f);
  }
}

// ---------------- Phase 6: Sinkhorn ----------------------------------------
__global__ void k_row(const float* __restrict__ Z, const float* __restrict__ vv,
                      float* __restrict__ uu){
  int i = blockIdx.x, b = blockIdx.y, t = threadIdx.x;
  const float* Zr = Z + (size_t)b*NAUG*NAUG + (size_t)i*NAUG;
  const float* vb = vv + b*NAUG;
  __shared__ float red[256];
  float mx = -INFINITY;
  for (int j = t; j < NAUG; j += 256) mx = fmaxf(mx, Zr[j] + vb[j]);
  red[t] = mx; __syncthreads();
  for (int s = 128; s > 0; s >>= 1){ if (t < s) red[t] = fmaxf(red[t], red[t+s]); __syncthreads(); }
  float M = red[0]; __syncthreads();
  float sum = 0.f;
  for (int j = t; j < NAUG; j += 256) sum += expf(Zr[j] + vb[j] - M);
  red[t] = sum; __syncthreads();
  for (int s = 128; s > 0; s >>= 1){ if (t < s) red[t] += red[t+s]; __syncthreads(); }
  if (t == 0){
    float lse = M + logf(red[0]);
    float norm = -logf((float)(KK + KK));
    float logmu = (i < NAUG-1) ? norm : (logf((float)KK) + norm);
    uu[b*NAUG + i] = logmu - lse;
  }
}

// thread-per-column online logsumexp: consecutive lanes read consecutive
// columns -> coalesced row-major traversal of Z. Prefetch 8 rows ahead
// (global_prefetch_b8) to stream Z out of L2.
__global__ void k_col(const float* __restrict__ Z, const float* __restrict__ uu,
                      float* __restrict__ vv){
  int j = blockIdx.x*blockDim.x + threadIdx.x;
  int b = blockIdx.y;
  if (j >= NAUG) return;
  const float* Zb = Z + (size_t)b*NAUG*NAUG;
  const float* ub = uu + b*NAUG;
  float M = -INFINITY, S = 0.f;
  for (int i = 0; i < NAUG; ++i){
    if (i + 8 < NAUG) __builtin_prefetch(Zb + (size_t)(i+8)*NAUG + j, 0, 0);
    float tv = Zb[(size_t)i*NAUG + j] + ub[i];
    if (tv > M){ S = S*expf(M - tv) + 1.f; M = tv; }
    else       { S += expf(tv - M); }
  }
  float lse = M + logf(S);
  float norm = -logf((float)(KK + KK));
  float lognu = (j < NAUG-1) ? norm : (logf((float)KK) + norm);
  vv[b*NAUG + j] = lognu - lse;
}

__global__ void k_final(float* __restrict__ Z, const float* __restrict__ uu,
                        const float* __restrict__ vv){
  long long gid = (long long)blockIdx.x*blockDim.x + threadIdx.x;
  long long total = (long long)BN*NAUG*NAUG;
  if (gid >= total) return;
  int b = (int)(gid / ((long long)NAUG*NAUG));
  long long rem = gid - (long long)b*NAUG*NAUG;
  int i = (int)(rem / NAUG), j = (int)(rem % NAUG);
  float norm = -logf((float)(KK + KK));
  Z[gid] = expf(Z[gid] + uu[b*NAUG + i] + vv[b*NAUG + j] - norm);
}

// ---------------------------------------------------------------------------
extern "C" void kernel_launch(void* const* d_in, const int* in_sizes, int n_in,
                              void* d_out, int out_size, void* d_ws, size_t ws_size,
                              hipStream_t stream){
  (void)in_sizes; (void)n_in; (void)out_size; (void)ws_size;
  const float* img1  = (const float*)d_in[0];
  const float* img2  = (const float*)d_in[1];
  const float* offsA = (const float*)d_in[2];
  const float* offsB = (const float*)d_in[3];

  float* out = (float*)d_out;
  float* kp1 = out;                       // BN*KK*2
  float* kp2 = out + BN*KK*2;             // BN*KK*2
  float* Z   = out + 2*BN*KK*2;           // BN*NAUG*NAUG (Z in-place -> probs)

  size_t off = 0;
  auto wsAlloc = [&](size_t bytes)->void*{
    void* p = (char*)d_ws + off;
    off += (bytes + 255) & ~(size_t)255;
    return p;
  };
  float*    sbuf    = (float*)   wsAlloc((size_t)BN*IH*IW*4);
  float*    smbuf   = (float*)   wsAlloc((size_t)BN*IH*IW*4);
  unsigned* hist    = (unsigned*)wsAlloc((size_t)BN*65536*4);
  int*      sel     = (int*)     wsAlloc((size_t)BN*2*4);
  unsigned* cnt     = (unsigned*)wsAlloc((size_t)BN*4);
  unsigned* cntT    = (unsigned*)wsAlloc((size_t)BN*4);
  float*    pairSc  = (float*)   wsAlloc((size_t)BN*KK*4);
  int*      pairIdx = (int*)     wsAlloc((size_t)BN*KK*4);
  float*    validB  = (float*)   wsAlloc((size_t)2*BN*KK*4);
  _Float16* dbuf    = (_Float16*)wsAlloc((size_t)2*BN*KK*PP*2);
  float*    sqbuf   = (float*)   wsAlloc((size_t)2*BN*KK*4);
  float*    uvec    = (float*)   wsAlloc((size_t)BN*NAUG*4);
  float*    vvec    = (float*)   wsAlloc((size_t)BN*NAUG*4);

  const int NPIX = BN*IH*IW;
  dim3 pixGrid((NPIX + 255)/256);

  for (int imgi = 0; imgi < 2; ++imgi){
    const float* img = imgi ? img2 : img1;
    float* kp = imgi ? kp2 : kp1;
    float* vb = validB + (size_t)imgi*BN*KK;
    k_shitomasi<<<pixGrid, 256, 0, stream>>>(img, sbuf);
    k_nms<<<pixGrid, 256, 0, stream>>>(sbuf, smbuf);
    k_zero<<<(BN*65536 + 255)/256, 256, 0, stream>>>(hist, BN*65536);
    k_zero<<<1, 256, 0, stream>>>(cnt, BN);
    k_zero<<<1, 256, 0, stream>>>(cntT, BN);
    k_hist<<<pixGrid, 256, 0, stream>>>(smbuf, hist);
    k_select<<<BN, 32, 0, stream>>>(hist, sel);
    k_compact<<<pixGrid, 256, 0, stream>>>(smbuf, sel, cnt, cntT, pairSc, pairIdx);
    k_sortemit<<<BN, 1024, 0, stream>>>(pairSc, pairIdx, cnt, kp, vb);
    k_desc<<<BN*KK, PP, 0, stream>>>(img, kp, vb, offsA, offsB,
                                     dbuf + (size_t)imgi*BN*KK*PP,
                                     sqbuf + (size_t)imgi*BN*KK);
  }

  k_auginit<<<dim3((NAUG + 255)/256, BN), 256, 0, stream>>>(Z);
  // 32x32 tile per wave, 4 waves per block -> grid (KK/128, KK/32, BN)
  k_gemm<<<dim3(KK/128, KK/32, BN), 128, 0, stream>>>(
      dbuf, dbuf + (size_t)BN*KK*PP, sqbuf, sqbuf + (size_t)BN*KK, Z);

  k_zero<<<(BN*NAUG + 255)/256, 256, 0, stream>>>((unsigned*)uvec, BN*NAUG);
  k_zero<<<(BN*NAUG + 255)/256, 256, 0, stream>>>((unsigned*)vvec, BN*NAUG);
  for (int it = 0; it < SINK_ITERS; ++it){
    k_row<<<dim3(NAUG, BN), 256, 0, stream>>>(Z, vvec, uvec);
    k_col<<<dim3((NAUG + 255)/256, BN), 256, 0, stream>>>(Z, uvec, vvec);
  }
  long long total = (long long)BN*NAUG*NAUG;
  k_final<<<(unsigned)((total + 255)/256), 256, 0, stream>>>(Z, uvec, vvec);
}
